// GNN_n1_80487687127055
// MI455X (gfx1250) — matrix operand order repaired and checked by reading, compiled, not run on hardware
//
#include <hip/hip_runtime.h>

typedef __attribute__((ext_vector_type(16))) _Float16 v16h;
typedef __attribute__((ext_vector_type(8)))  _Float16 v8h;
typedef __attribute__((ext_vector_type(8)))  float    v8f;

namespace {

constexpr int kB    = 4;
constexpr int kN    = 192;
constexpr int kF0   = 128;
constexpr int kH    = 48;     // NF/2
constexpr int kNOut = 64;
constexpr int kLdx  = 224;    // final feature width (x0|l1|l2)
constexpr int kM    = kB * kN * kN;  // 147456 pair rows
constexpr int kBN   = kB * kN;       // 768 node rows
constexpr int kRedBlocks = 256;      // blocks in BN partial reduction
constexpr int kRedRows   = kM / kRedBlocks;  // 576 rows per block

// ---------------- workspace map (bytes) ----------------
constexpr size_t OFF_D   = 0;                           // f16 [M,224] max pair-diff
constexpr size_t SZ_D    = (size_t)kM * 224 * 2;
constexpr size_t OFF_H1  = OFF_D + SZ_D;                // f16 [M,192]
constexpr size_t SZ_H    = (size_t)kM * 192 * 2;
constexpr size_t OFF_H2  = OFF_H1 + SZ_H;               // f16 [M,192]
constexpr size_t OFF_LOG = OFF_H2 + SZ_H;               // f32 [M]
constexpr size_t SZ_LOG  = (size_t)kM * 4;
constexpr size_t OFF_A0  = OFF_LOG + SZ_LOG;            // f32 [B,N,N] softmax (stage1)
constexpr size_t OFF_AL  = OFF_A0 + SZ_LOG;             // f32 [B,N,N] softmax (stage3)
constexpr size_t OFF_X   = OFF_AL + SZ_LOG;             // f32 [B,N,224] features
constexpr size_t SZ_X    = (size_t)kBN * kLdx * 4;
constexpr size_t OFF_Y2  = OFF_X + SZ_X;                // f32 [B,N,224] A@x
constexpr size_t OFF_G   = OFF_Y2 + SZ_X;               // f32 [B*N,48] pre-BN linear
constexpr size_t SZ_G    = (size_t)kBN * kH * 4;
constexpr size_t OFF_SC  = OFF_G + SZ_G;                // f32[256] BN scale
constexpr size_t OFF_SH  = OFF_SC + 1024;               // f32[256] BN shift
constexpr size_t OFF_P0  = OFF_SH + 1024;               // f32 [256 blocks][256] partial sums
constexpr size_t SZ_P    = (size_t)kRedBlocks * 256 * 4;
constexpr size_t OFF_P1  = OFF_P0 + SZ_P;               // f32 [256 blocks][256] partial sumsq
constexpr size_t OFF_W16 = OFF_P1 + SZ_P;               // f16 converted weights
constexpr size_t WS_NEED = OFF_W16 + 512 * 1024;

// f16 weight sub-offsets (in halves, all 32B aligned)
constexpr size_t W1C1 = 0;
constexpr size_t W1C2 = W1C1 + 192 * 128;
constexpr size_t W1C3 = W1C2 + 192 * 192;
constexpr size_t W1C4 = W1C3 + 96 * 192;
constexpr size_t WLC1 = W1C4 + 96 * 96;
constexpr size_t WLC2 = WLC1 + 192 * 224;
constexpr size_t WLC3 = WLC2 + 192 * 192;
constexpr size_t WLC4 = WLC3 + 96 * 192;

// ---------------- kernels ----------------

__global__ void cvt16_k(_Float16* __restrict__ dst, const float* __restrict__ src, int n) {
  int i = blockIdx.x * blockDim.x + threadIdx.x;
  if (i < n) dst[i] = (_Float16)src[i];
}

__global__ void init_x_k(float* __restrict__ X, const float* __restrict__ xin,
                         int rows, int F, int ldx) {
  int i = blockIdx.x * blockDim.x + threadIdx.x;
  if (i >= rows * F) return;
  int r = i / F, f = i - r * F;
  X[(size_t)r * ldx + f] = xin[i];
}

// D[m, f] = |x_i - x_j|  (m = (b*N+i)*N + j), f16 output, row stride F
__global__ void pairdiff_k(_Float16* __restrict__ D, const float* __restrict__ X,
                           int N, int F, int ldx) {
  int m = blockIdx.x;
  int j = m % N;
  int bi = m / N;
  const float* xi = X + (size_t)bi * ldx;
  const float* xj = X + (size_t)((bi / N) * N + j) * ldx;
  _Float16* drow = D + (size_t)m * F;
  for (int f = threadIdx.x; f < F; f += blockDim.x)
    drow[f] = (_Float16)fabsf(xi[f] - xj[f]);
}

// C[M, 16*NT] = A[M,K] * W[16*NT, K]^T + bias  (all-f16 in, f16 out, f32 accum via WMMA)
// 8 waves per block, one 16-row strip per wave, wave owns all NT column tiles.
template <int NT>
__global__ __launch_bounds__(256) void gemm_nt_f16(
    _Float16* __restrict__ C, const _Float16* __restrict__ A,
    const _Float16* __restrict__ W, const float* __restrict__ bias, int K) {
  constexpr int Nc = NT * 16;
  const int lane  = threadIdx.x & 31;
  const int wave  = threadIdx.x >> 5;
  const int mbase = (blockIdx.x * 8 + wave) * 16;
  const int lr = lane & 15;
  const int hs = lane >> 4;  // lane-half select
  v8f zero = {};
  v8f acc[NT];
#pragma unroll
  for (int t = 0; t < NT; ++t) acc[t] = zero;
  const _Float16* arow = A + (size_t)(mbase + lr) * K;
  const int kb = hs * 16;
  for (int k0 = 0; k0 < K; k0 += 32) {
    // A frag: lanes 0-15 row lr holds K[k0..k0+15]; lanes 16-31 K[k0+16..k0+31]
    const _Float16* ap = arow + k0 + kb;
    v16h af = *(const v16h*)ap;
#pragma unroll
    for (int t = 0; t < NT; ++t) {
      // B frag: lane holds 16 contiguous K of weight-row (= B column) t*16+lr
      const _Float16* bp = W + (size_t)(t * 16 + lr) * K + k0 + hs * 16;
      v16h bf = *(const v16h*)bp;
      acc[t] = __builtin_amdgcn_wmma_f32_16x16x32_f16(false, af, false, bf,
                                                      (short)0, acc[t], false, false);
    }
  }
#pragma unroll
  for (int t = 0; t < NT; ++t) {
    const int col = t * 16 + lr;
    const float bv = bias[col];
#pragma unroll
    for (int r = 0; r < 8; ++r) {
      const int row = mbase + hs * 8 + r;  // v_r: lanes0-15 -> M=r, lanes16-31 -> M=r+8
      C[(size_t)row * Nc + col] = (_Float16)(acc[t][r] + bv);
    }
  }
}

// ---- BN stats, coalesced two-phase deterministic reduction (f16 activations) ----
// Phase 1: block owns rows [blk*R, (blk+1)*R); thread c (blockDim==Nc) walks column c
// with row-major (coalesced) reads; writes per-block partials.
__global__ void bn_partial_k(const _Float16* __restrict__ H, int rowsPerBlock,
                             float* __restrict__ part0, float* __restrict__ part1) {
  const int Nc = blockDim.x;
  const int c = threadIdx.x;
  const _Float16* p = H + (size_t)blockIdx.x * rowsPerBlock * Nc + c;
  float s = 0.f, ss = 0.f;
  for (int r = 0; r < rowsPerBlock; ++r) {
    float v = (float)p[(size_t)r * Nc];
    s += v;
    ss += v * v;
  }
  part0[(size_t)blockIdx.x * 256 + c] = s;
  part1[(size_t)blockIdx.x * 256 + c] = ss;
}

// Phase 2: combine partials per channel, fold gamma/beta into scale/shift.
__global__ void bn_finalize_k(const float* __restrict__ part0, const float* __restrict__ part1,
                              int nblocks, int M, int Nc,
                              const float* __restrict__ gamma, const float* __restrict__ beta,
                              float* __restrict__ scale, float* __restrict__ shift) {
  int c = blockIdx.x * blockDim.x + threadIdx.x;
  if (c >= Nc) return;
  float s = 0.f, ss = 0.f;
  for (int b = 0; b < nblocks; ++b) {
    s += part0[(size_t)b * 256 + c];
    ss += part1[(size_t)b * 256 + c];
  }
  float mean = s / (float)M;
  float var = ss / (float)M - mean * mean;  // biased, matches jnp.var
  float inv = rsqrtf(var + 1e-5f);
  float sc = gamma[c] * inv;
  scale[c] = sc;
  shift[c] = beta[c] - mean * sc;
}

// small f32 BN stats (768 rows x 48 cols), one block per channel
__global__ __launch_bounds__(256) void bn_stats_f32_k(const float* __restrict__ H, int M, int Nc,
                                                      const float* __restrict__ gamma,
                                                      const float* __restrict__ beta,
                                                      float* __restrict__ scale,
                                                      float* __restrict__ shift) {
  int c = blockIdx.x;
  float s = 0.f, ss = 0.f;
  for (int m = threadIdx.x; m < M; m += 256) {
    float v = H[(size_t)m * Nc + c];
    s += v;
    ss += v * v;
  }
  __shared__ float sh0[256], sh1[256];
  sh0[threadIdx.x] = s;
  sh1[threadIdx.x] = ss;
  __syncthreads();
  for (int off = 128; off > 0; off >>= 1) {
    if (threadIdx.x < off) {
      sh0[threadIdx.x] += sh0[threadIdx.x + off];
      sh1[threadIdx.x] += sh1[threadIdx.x + off];
    }
    __syncthreads();
  }
  if (threadIdx.x == 0) {
    float mean = sh0[0] / (float)M;
    float var = sh1[0] / (float)M - mean * mean;
    float inv = rsqrtf(var + 1e-5f);
    float sc = gamma[c] * inv;
    scale[c] = sc;
    shift[c] = beta[c] - mean * sc;
  }
}

// in-place BN + leaky-relu on f16 activations
__global__ void bn_apply_f16_k(_Float16* __restrict__ H, size_t total, int Nc,
                               const float* __restrict__ scale,
                               const float* __restrict__ shift) {
  size_t i = (size_t)blockIdx.x * blockDim.x + threadIdx.x;
  if (i >= total) return;
  int c = (int)(i % (size_t)Nc);
  float v = (float)H[i] * scale[c] + shift[c];
  H[i] = (_Float16)(v > 0.f ? v : 0.01f * v);
}

// logits[m] = H4[m,:96] . clw + clb
__global__ void logits_k(float* __restrict__ logits, const _Float16* __restrict__ H,
                         const float* __restrict__ clw, const float* __restrict__ clb,
                         int M, int K) {
  int m = blockIdx.x * blockDim.x + threadIdx.x;
  if (m >= M) return;
  const _Float16* h = H + (size_t)m * K;
  float acc = clb[0];
  for (int k = 0; k < K; ++k) acc += (float)h[k] * clw[k];
  logits[m] = acc;
}

// row softmax with -1e8 diagonal mask
__global__ __launch_bounds__(256) void softmax_k(float* __restrict__ Aout,
                                                 const float* __restrict__ logits, int N) {
  int row = blockIdx.x;  // b*N + i
  int i = row % N;
  int t = threadIdx.x;
  const float* lr = logits + (size_t)row * N;
  float v = -3.0e38f;
  if (t < N) {
    v = lr[t];
    if (t == i) v -= 1e8f;
  }
  __shared__ float sm[256];
  sm[t] = v;
  __syncthreads();
  for (int off = 128; off > 0; off >>= 1) {
    if (t < off) sm[t] = fmaxf(sm[t], sm[t + off]);
    __syncthreads();
  }
  float mx = sm[0];
  __syncthreads();
  float e = (t < N) ? __expf(v - mx) : 0.f;
  sm[t] = e;
  __syncthreads();
  for (int off = 128; off > 0; off >>= 1) {
    if (t < off) sm[t] += sm[t + off];
    __syncthreads();
  }
  if (t < N) Aout[(size_t)row * N + t] = e / sm[0];
}

// Y2[b,i,f] = sum_j A[b,i,j] * X[b,j,f]
__global__ void gmul_k(float* __restrict__ Y2, const float* __restrict__ Aw,
                       const float* __restrict__ X, int N, int F, int ldx) {
  int row = blockIdx.x;  // b*N + i
  int b = row / N;
  const float* arow = Aw + (size_t)row * N;
  for (int f = threadIdx.x; f < F; f += blockDim.x) {
    float acc = 0.f;
    const float* xb = X + (size_t)b * N * ldx + f;
    for (int j = 0; j < N; ++j) acc += arow[j] * xb[(size_t)j * ldx];
    Y2[(size_t)row * ldx + f] = acc;
  }
}

// G[m,o] = concat(X[m,:F], Y2[m,:F]) . Wl[o,:] + bl[o]
__global__ void linear_k(float* __restrict__ G, const float* __restrict__ X,
                         const float* __restrict__ Y2, const float* __restrict__ Wl,
                         const float* __restrict__ bl, int F, int ldx, int H) {
  int m = blockIdx.x;
  int o = threadIdx.x;
  if (o >= H) return;
  const float* w = Wl + (size_t)o * 2 * F;
  const float* xr = X + (size_t)m * ldx;
  const float* yr = Y2 + (size_t)m * ldx;
  float acc = bl[o];
  for (int k = 0; k < F; ++k) acc += xr[k] * w[k];
  for (int k = 0; k < F; ++k) acc += yr[k] * w[F + k];
  G[(size_t)m * H + o] = acc;
}

// BN + lrelu on G, concatenated into X at column offset dstoff
__global__ void bn_apply_concat_k(float* __restrict__ X, const float* __restrict__ G,
                                  int M, int H, int ldx, int dstoff,
                                  const float* __restrict__ scale,
                                  const float* __restrict__ shift) {
  int i = blockIdx.x * blockDim.x + threadIdx.x;
  if (i >= M * H) return;
  int m = i / H, c = i - m * H;
  float v = G[i] * scale[c] + shift[c];
  v = v > 0.f ? v : 0.01f * v;
  X[(size_t)m * ldx + dstoff + c] = v;
}

// out[b,o] = concat(X[b,0,:F], sum_j Al[b,0,j] X[b,j,:F]) . Wll[o,:] + bll[o]
__global__ __launch_bounds__(256) void final_k(float* __restrict__ out,
                                               const float* __restrict__ Al,
                                               const float* __restrict__ X,
                                               const float* __restrict__ Wll,
                                               const float* __restrict__ bll,
                                               int N, int F, int ldx, int NOUT) {
  int b = blockIdx.x;
  __shared__ float ysh[kLdx];
  const float* arow = Al + (size_t)b * N * N;  // row i=0
  for (int f = threadIdx.x; f < F; f += blockDim.x) {
    float acc = 0.f;
    const float* xb = X + (size_t)b * N * ldx + f;
    for (int j = 0; j < N; ++j) acc += arow[j] * xb[(size_t)j * ldx];
    ysh[f] = acc;
  }
  __syncthreads();
  const float* x0 = X + (size_t)b * N * ldx;  // node 0
  for (int o = threadIdx.x; o < NOUT; o += blockDim.x) {
    const float* w = Wll + (size_t)o * 2 * F;
    float acc = bll[o];
    for (int k = 0; k < F; ++k) acc += x0[k] * w[k];
    for (int k = 0; k < F; ++k) acc += ysh[k] * w[F + k];
    out[b * NOUT + o] = acc;
  }
}

// ---------------- host-side pipeline helpers ----------------

void run_bn_f16(hipStream_t stream, char* ws, _Float16* H, int Nc,
                const float* gamma, const float* beta) {
  float* part0 = (float*)(ws + OFF_P0);
  float* part1 = (float*)(ws + OFF_P1);
  float* scale = (float*)(ws + OFF_SC);
  float* shift = (float*)(ws + OFF_SH);
  bn_partial_k<<<kRedBlocks, Nc, 0, stream>>>(H, kRedRows, part0, part1);
  bn_finalize_k<<<1, 256, 0, stream>>>(part0, part1, kRedBlocks, kM, Nc, gamma, beta,
                                       scale, shift);
  bn_apply_f16_k<<<(unsigned)(((size_t)kM * Nc + 255) / 256), 256, 0, stream>>>(
      H, (size_t)kM * Nc, Nc, scale, shift);
}

// d_in[pbase + k]: 0:c1 1:c1b 2:g1 3:b1 4:c2 5:c2b 6:g2 7:b2 8:c3 9:c3b 10:g3 11:b3
//                  12:c4 13:c4b 14:g4 15:b4 16:cl 17:clb
void run_wcompute(hipStream_t stream, void* const* d_in, int pbase, char* ws,
                  const float* X, int F, const _Float16* wc1, const _Float16* wc2,
                  const _Float16* wc3, const _Float16* wc4, float* Aout) {
  _Float16* D  = (_Float16*)(ws + OFF_D);
  _Float16* H1 = (_Float16*)(ws + OFF_H1);
  _Float16* H2 = (_Float16*)(ws + OFF_H2);
  float* logits = (float*)(ws + OFF_LOG);
  auto P = [&](int k) { return (const float*)d_in[pbase + k]; };

  pairdiff_k<<<kM, 128, 0, stream>>>(D, X, kN, F, kLdx);

  gemm_nt_f16<12><<<kM / 128, 256, 0, stream>>>(H1, D, wc1, P(1), F);
  run_bn_f16(stream, ws, H1, 192, P(2), P(3));

  gemm_nt_f16<12><<<kM / 128, 256, 0, stream>>>(H2, H1, wc2, P(5), 192);
  run_bn_f16(stream, ws, H2, 192, P(6), P(7));

  gemm_nt_f16<6><<<kM / 128, 256, 0, stream>>>(H1, H2, wc3, P(9), 192);
  run_bn_f16(stream, ws, H1, 96, P(10), P(11));

  gemm_nt_f16<6><<<kM / 128, 256, 0, stream>>>(H2, H1, wc4, P(13), 96);
  run_bn_f16(stream, ws, H2, 96, P(14), P(15));

  logits_k<<<(kM + 255) / 256, 256, 0, stream>>>(logits, H2, P(16), P(17), kM, 96);
  softmax_k<<<kBN, 256, 0, stream>>>(Aout, logits, kN);
}

}  // namespace

// d_in order (recursive dict order): 0:x, 1-18:w1{c1,c1b,g1,b1,c2,c2b,g2,b2,c3,c3b,g3,b3,
// c4,c4b,g4,b4,cl,clb}, 19-36:w2(dead), 37-54:wl, 55-58:l1{W,b,g,be}, 59-62:l2, 63-64:ll{W,b}
extern "C" void kernel_launch(void* const* d_in, const int* in_sizes, int n_in,
                              void* d_out, int out_size, void* d_ws, size_t ws_size,
                              hipStream_t stream) {
  (void)in_sizes; (void)n_in; (void)out_size;
  if (ws_size < WS_NEED) return;
  char* ws = (char*)d_ws;
  float* X     = (float*)(ws + OFF_X);
  float* Y2    = (float*)(ws + OFF_Y2);
  float* G     = (float*)(ws + OFF_G);
  float* A0    = (float*)(ws + OFF_A0);
  float* AL    = (float*)(ws + OFF_AL);
  float* scale = (float*)(ws + OFF_SC);
  float* shift = (float*)(ws + OFF_SH);
  _Float16* w16 = (_Float16*)(ws + OFF_W16);

  auto cv = [&](size_t off, int idx, int n) {
    cvt16_k<<<(n + 255) / 256, 256, 0, stream>>>(w16 + off, (const float*)d_in[idx], n);
  };
  cv(W1C1, 1, 192 * 128);
  cv(W1C2, 5, 192 * 192);
  cv(W1C3, 9, 96 * 192);
  cv(W1C4, 13, 96 * 96);
  cv(WLC1, 37, 192 * 224);
  cv(WLC2, 41, 192 * 192);
  cv(WLC3, 45, 96 * 192);
  cv(WLC4, 49, 96 * 96);

  init_x_k<<<(kBN * kF0 + 255) / 256, 256, 0, stream>>>(X, (const float*)d_in[0], kBN, kF0, kLdx);

  // stage 1: W0 = wcompute(x0, w1), F=128
  run_wcompute(stream, d_in, 1, ws, X, kF0, w16 + W1C1, w16 + W1C2, w16 + W1C3, w16 + W1C4, A0);

  // gconv l1 (uses W0, F=128) -> X cols [128,176)
  gmul_k<<<kBN, 256, 0, stream>>>(Y2, A0, X, kN, kF0, kLdx);
  linear_k<<<kBN, 64, 0, stream>>>(G, X, Y2, (const float*)d_in[55], (const float*)d_in[56],
                                   kF0, kLdx, kH);
  bn_stats_f32_k<<<kH, 256, 0, stream>>>(G, kBN, kH, (const float*)d_in[57],
                                         (const float*)d_in[58], scale, shift);
  bn_apply_concat_k<<<(kBN * kH + 255) / 256, 256, 0, stream>>>(X, G, kBN, kH, kLdx, kF0,
                                                                scale, shift);

  // (w2 wcompute is dead code in the reference -> skipped)

  // gconv l2 (uses W0 again, F=176) -> X cols [176,224)
  gmul_k<<<kBN, 256, 0, stream>>>(Y2, A0, X, kN, 176, kLdx);
  linear_k<<<kBN, 64, 0, stream>>>(G, X, Y2, (const float*)d_in[59], (const float*)d_in[60],
                                   176, kLdx, kH);
  bn_stats_f32_k<<<kH, 256, 0, stream>>>(G, kBN, kH, (const float*)d_in[61],
                                         (const float*)d_in[62], scale, shift);
  bn_apply_concat_k<<<(kBN * kH + 255) / 256, 256, 0, stream>>>(X, G, kBN, kH, kLdx, 176,
                                                                scale, shift);

  // stage 3: Wl = wcompute(x2, wl), F=224
  run_wcompute(stream, d_in, 37, ws, X, kLdx, w16 + WLC1, w16 + WLC2, w16 + WLC3, w16 + WLC4, AL);

  // final gconv (bn=False), only node row 0 needed
  final_k<<<kB, 256, 0, stream>>>((float*)d_out, AL, X, (const float*)d_in[63],
                                  (const float*)d_in[64], kN, kLdx, kLdx, kNOut);
}
